// AttentionMechanism_10986526343496
// MI455X (gfx1250) — matrix operand order repaired and checked by reading, not compiled, not run on hardware
//
#include <hip/hip_runtime.h>
#include <hip/hip_bf16.h>

typedef __attribute__((ext_vector_type(16))) __bf16 bf16x16;
typedef __attribute__((ext_vector_type(8)))  float  f32x8;
typedef __attribute__((ext_vector_type(4)))  unsigned int u32x4;
typedef __attribute__((ext_vector_type(8)))  int i32x8;
typedef __attribute__((ext_vector_type(4)))  int i32x4;

#define B_   2
#define S_   2048
#define D_   1024
#define N3_  3072
#define H_   16
#define DH_  64
#define QKV_ELEMS (B_ * H_ * S_ * DH_)   // 4194304 per tensor
#define XB_ELEMS  (B_ * S_ * D_)         // 4194304
#define WT_ELEMS  (N3_ * D_)             // 3145728

// Native bf16 convert (backend picks v_cvt_(pk_)bf16_f32)
__device__ __forceinline__ unsigned short f2bf(float f) {
  union { __bf16 h; unsigned short u; } c;
  c.h = (__bf16)f;
  return c.u;
}
__device__ __forceinline__ unsigned pack2bf(float a, float b) {
  union { __bf16 h[2]; unsigned u; } c;
  c.h[0] = (__bf16)a; c.h[1] = (__bf16)b;
  return c.u;
}

// Generic->LDS offset: LDS aperture keeps the byte offset in addr[31:0]
__device__ __forceinline__ unsigned lds_off(const void* p) {
  return (unsigned)(unsigned long long)p;
}

// CDNA5 async global->LDS copy, 16B per lane, GVS addressing (saddr + 32-bit voffset)
__device__ __forceinline__ void async_b128(unsigned lds_addr, unsigned goff_bytes, const void* sbase) {
  asm volatile("global_load_async_to_lds_b128 %0, %1, %2"
               :: "v"(lds_addr), "v"(goff_bytes), "s"(sbase) : "memory");
}

// CDNA5 Tensor Data Mover: 2D tile (elements are 2 bytes), groups per ISA cdna5 ch.8.
// tile_d0 = elements per row, tile_d1 = rows, stride0 = global row stride in elements.
// LDS receives rows packed contiguously (tile_d0*2 bytes per row).
__device__ __forceinline__ void tdm_load_2d(unsigned lds_addr, const void* gaddr,
                                            unsigned tile_d0, unsigned tile_d1,
                                            unsigned tensor_d0, unsigned tensor_d1,
                                            unsigned long long stride0_elems) {
  unsigned long long ga = (unsigned long long)gaddr;
  u32x4 g0;
  g0[0] = 1u;                                   // count=1, user descriptor
  g0[1] = lds_addr;                             // [63:32] lds_addr
  g0[2] = (unsigned)ga;                         // [95:64] global_addr lo
  g0[3] = (unsigned)(ga >> 32) | (2u << 30);    // [120:96] global_addr hi, [127:126] type=2
  i32x8 g1;
  g1[0] = 0x00010000;                           // workgroup_mask=0, data_size=1 (2B)
  g1[1] = (int)(tensor_d0 << 16);               // [63:48] tensor_dim0 lo16
  g1[2] = (int)((tensor_d0 >> 16) | (tensor_d1 << 16));  // [79:64] dim0 hi, [95:80] dim1 lo
  g1[3] = (int)((tensor_d1 >> 16) | (tile_d0 << 16));    // [111:96] dim1 hi, [127:112] tile_dim0
  g1[4] = (int)(tile_d1 & 0xFFFFu);             // [143:128] tile_dim1, [159:144] tile_dim2=0
  g1[5] = (int)(unsigned)(stride0_elems & 0xFFFFFFFFull);        // [191:160] dim0_stride lo
  g1[6] = (int)(unsigned)((stride0_elems >> 32) & 0xFFFFull);    // [207:192] dim0_stride hi
  g1[7] = 0;
  i32x4 z4 = {0, 0, 0, 0};
  i32x8 z8 = {0, 0, 0, 0, 0, 0, 0, 0};
  // amdgpu-toolchain (clang-23 / therock-10.0) 6-arg form
  __builtin_amdgcn_tensor_load_to_lds(g0, g1, z4, z4, z8, 0);
}

// A-fragment (16x32 bf16): lane holds row = lane%16; element i -> K = (i/8)*16 + (lane/16)*8 + (i%8)
__device__ __forceinline__ bf16x16 load_fragA(const unsigned short* row_ptr) {
  int half = (threadIdx.x >> 4) & 1;
  union { bf16x16 v; uint4 q[2]; } f;
  f.q[0] = *(const uint4*)(row_ptr + half * 8);
  f.q[1] = *(const uint4*)(row_ptr + 16 + half * 8);
  return f.v;
}
// B-fragment (32x16 bf16): lane holds column = lane%16; element i -> K = (i/8)*16 + (lane/16)*8 + (i%8)
__device__ __forceinline__ bf16x16 load_fragB(const unsigned short* col_ptr) {
  int half = (threadIdx.x >> 4) & 1;
  union { bf16x16 v; uint4 q[2]; } f;
  f.q[0] = *(const uint4*)(col_ptr + half * 8);
  f.q[1] = *(const uint4*)(col_ptr + 16 + half * 8);
  return f.v;
}
__device__ __forceinline__ f32x8 wmma_bf16(bf16x16 a, bf16x16 b, f32x8 c) {
  return __builtin_amdgcn_wmma_f32_16x16x32_bf16(false, a, false, b, (short)0, c, false, false);
}

// ---------------- Kernel 0: one-pass prep ----------------
__global__ __launch_bounds__(256) void prep_kernel(
    const float* __restrict__ X, const float* __restrict__ W,
    unsigned short* __restrict__ Xb, unsigned short* __restrict__ Wt)
{
  if (blockIdx.x < 4096) {
    int e0 = (blockIdx.x * 256 + threadIdx.x) * 4;          // over 4194304
    float4 v = *(const float4*)(X + e0);
    uint2 p; p.x = pack2bf(v.x, v.y); p.y = pack2bf(v.z, v.w);
    *(uint2*)(Xb + e0) = p;
  } else {
    int c = (blockIdx.x - 4096) * 256 + threadIdx.x;        // over 786432
    int k = c / 768;
    int n = (c - k * 768) * 4;
    float4 v = *(const float4*)(W + (size_t)k * N3_ + n);
    Wt[(size_t)(n + 0) * D_ + k] = f2bf(v.x);
    Wt[(size_t)(n + 1) * D_ + k] = f2bf(v.y);
    Wt[(size_t)(n + 2) * D_ + k] = f2bf(v.z);
    Wt[(size_t)(n + 3) * D_ + k] = f2bf(v.w);
  }
}

// ---------------- Kernel 1: QKV GEMM (async-staged, double-buffered) ----------------
__global__ __launch_bounds__(256) void qkv_gemm_kernel(
    const unsigned short* __restrict__ Xb, const unsigned short* __restrict__ Wt,
    const float* __restrict__ bias,
    unsigned short* __restrict__ Qb, unsigned short* __restrict__ Kb,
    unsigned short* __restrict__ Vt)
{
  __shared__ unsigned short Xs[2][128 * 32];
  __shared__ unsigned short Ws[2][128 * 32];

  const int tid  = threadIdx.x;
  const int lane = tid & 31, wave = tid >> 5;
  const int ln16 = lane & 15, half = lane >> 4;
  const int wm = (wave >> 1) * 32, wn = (wave & 1) * 64;
  const int m0 = blockIdx.y * 128, n0 = blockIdx.x * 128;

  const unsigned short* xbase = Xb + (size_t)m0 * D_;
  const unsigned short* wbase = Wt + (size_t)n0 * D_;
  unsigned xs[2] = { lds_off(Xs[0]), lds_off(Xs[1]) };
  unsigned ws[2] = { lds_off(Ws[0]), lds_off(Ws[1]) };

  auto stage = [&](int buf, int k0) {
#pragma unroll
    for (int i = 0; i < 2; ++i) {
      int t = tid + i * 256;                 // 0..511
      int row = t >> 2, ch = (t & 3) * 16;
      unsigned goff = (unsigned)row * (D_ * 2) + (unsigned)k0 * 2 + ch;
      async_b128(xs[buf] + row * 64 + ch, goff, xbase);
      async_b128(ws[buf] + row * 64 + ch, goff, wbase);
    }
  };

  const f32x8 zero = {0.f,0.f,0.f,0.f,0.f,0.f,0.f,0.f};
  f32x8 acc[2][4];
#pragma unroll
  for (int i = 0; i < 2; ++i)
#pragma unroll
    for (int j = 0; j < 4; ++j) acc[i][j] = zero;

  stage(0, 0);
  for (int it = 0; it < D_ / 32; ++it) {
    const int buf = it & 1;
    __syncthreads();
    if (it + 1 < D_ / 32) {
      stage(buf ^ 1, (it + 1) * 32);
      asm volatile("s_wait_asynccnt 0x4" ::: "memory");
    } else {
      asm volatile("s_wait_asynccnt 0x0" ::: "memory");
    }
    __syncthreads();

    const unsigned short* xlds = Xs[buf];
    const unsigned short* wlds = Ws[buf];
    bf16x16 a[2];
#pragma unroll
    for (int mt = 0; mt < 2; ++mt)
      a[mt] = load_fragA(&xlds[(wm + mt * 16 + ln16) * 32]);
#pragma unroll
    for (int nt = 0; nt < 4; ++nt) {
      bf16x16 bf = load_fragB(&wlds[(wn + nt * 16 + ln16) * 32]);
#pragma unroll
      for (int mt = 0; mt < 2; ++mt)
        acc[mt][nt] = wmma_bf16(a[mt], bf, acc[mt][nt]);
    }
  }

  // Epilogue: +bias; Q,K -> [b][h][s][d]; V -> [b][h][d][s]
#pragma unroll
  for (int nt = 0; nt < 4; ++nt) {
    int n = n0 + wn + nt * 16 + ln16;
    float bv = bias[n];
    int which = n >> 10;
    int rem = n & 1023;
    int h = rem >> 6, d = rem & 63;
#pragma unroll
    for (int mt = 0; mt < 2; ++mt) {
#pragma unroll
      for (int r = 0; r < 8; ++r) {
        int m = m0 + wm + mt * 16 + half * 8 + r;
        int b = m >> 11, s = m & (S_ - 1);
        unsigned short val = f2bf(acc[mt][nt][r] + bv);
        if (which == 2) {
          Vt[((size_t)((b * H_ + h) * DH_ + d)) * S_ + s] = val;
        } else {
          unsigned short* dst = which ? Kb : Qb;
          dst[((size_t)((b * H_ + h) * S_ + s)) * DH_ + d] = val;
        }
      }
    }
  }
}

// ---------------- Kernel 2: flash attention (TDM-staged K/V tiles) ----------------
// grid (S/64, H, B); 128 threads = 4 waves; each wave owns 16 query rows.
__global__ __launch_bounds__(128) void flash_attn_kernel(
    const unsigned short* __restrict__ Qb, const unsigned short* __restrict__ Kb,
    const unsigned short* __restrict__ Vt, float* __restrict__ out)
{
  __shared__ unsigned short Ks[2][32 * 64];    // [key][d]
  __shared__ unsigned short Vs[2][64 * 32];    // [d][key]
  __shared__ unsigned short Pw[4][16 * 32];    // per-wave P tile

  const int tid  = threadIdx.x;
  const int lane = tid & 31, wave = tid >> 5;
  const int ln16 = lane & 15, half = lane >> 4;
  const int b = blockIdx.z, h = blockIdx.y;
  const size_t bh = (size_t)(b * H_ + h) * S_;
  const int q0 = blockIdx.x * 64 + wave * 16;

  const unsigned short* kbase = Kb + bh * DH_;           // [key][d], tile contiguous
  const unsigned short* vbase = Vt + bh * DH_;           // [d][s], row stride S_
  unsigned ks[2] = { lds_off(Ks[0]), lds_off(Ks[1]) };
  unsigned vs[2] = { lds_off(Vs[0]), lds_off(Vs[1]) };

  // Wave 0 issues two TDM descriptors per tile (EXEC ignored for TENSOR ops):
  //  K: one 2048-element contiguous row; V: 64 rows x 32 elements, global row stride S_.
  auto stage = [&](int buf, int kb) {
    if (wave == 0) {
      tdm_load_2d(ks[buf], kbase + (size_t)kb * DH_, /*tile*/ 2048, 1,
                  /*tensor*/ (unsigned)(S_ * DH_), 1, (unsigned long long)(S_ * DH_));
      tdm_load_2d(vs[buf], vbase + kb, /*tile*/ 32, 64,
                  /*tensor*/ (unsigned)S_, (unsigned)DH_, (unsigned long long)S_);
    }
  };

  // Q fragments loaded once from global
  const unsigned short* qp = Qb + (bh + q0 + ln16) * DH_;
  bf16x16 qa[2];
  qa[0] = load_fragA(qp);
  qa[1] = load_fragA(qp + 32);

  const float cs = 0.125f * 1.44269504088896f;   // 1/sqrt(64) * log2(e)
  const f32x8 zero = {0.f,0.f,0.f,0.f,0.f,0.f,0.f,0.f};
  f32x8 o[4];
  o[0] = o[1] = o[2] = o[3] = zero;
  float mrow[8], lrow[8];
#pragma unroll
  for (int r = 0; r < 8; ++r) { mrow[r] = -__builtin_inff(); lrow[r] = 0.f; }

  stage(0, 0);
  for (int it = 0; it < S_ / 32; ++it) {
    const int buf = it & 1;
    __syncthreads();                             // everyone done reading buf^1
    if (it + 1 < S_ / 32) {
      stage(buf ^ 1, (it + 1) * 32);             // wave0: 2 TDM ops outstanding -> 4
      if (wave == 0) __builtin_amdgcn_s_wait_tensorcnt(2);   // tile `it` resident
    } else {
      if (wave == 0) __builtin_amdgcn_s_wait_tensorcnt(0);
    }
    __syncthreads();                             // publish tile `it` to all waves

    const unsigned short* Kt = Ks[buf];
    const unsigned short* Vtl = Vs[buf];

    // Scores over two 16-key tiles; K-dim = d (2 chunks of 32)
    f32x8 s0 = zero, s1 = zero;
#pragma unroll
    for (int c = 0; c < 2; ++c) {
      bf16x16 b0 = load_fragB(&Kt[ln16 * 64 + c * 32]);
      bf16x16 b1 = load_fragB(&Kt[(16 + ln16) * 64 + c * 32]);
      s0 = wmma_bf16(qa[c], b0, s0);
      s1 = wmma_bf16(qa[c], b1, s1);
    }

    // Online softmax (rows m = half*8 + r; reduce across 16-lane halves)
    float pr0[8], pr1[8], alpha[8];
#pragma unroll
    for (int r = 0; r < 8; ++r) {
      float x0 = s0[r] * cs, x1 = s1[r] * cs;
      float mx = fmaxf(x0, x1);
      mx = fmaxf(mx, __shfl_xor(mx, 1, 32));
      mx = fmaxf(mx, __shfl_xor(mx, 2, 32));
      mx = fmaxf(mx, __shfl_xor(mx, 4, 32));
      mx = fmaxf(mx, __shfl_xor(mx, 8, 32));
      float mnew = fmaxf(mrow[r], mx);
      float al = __builtin_exp2f(mrow[r] - mnew);
      float p0 = __builtin_exp2f(x0 - mnew);
      float p1 = __builtin_exp2f(x1 - mnew);
      float rs = p0 + p1;
      rs += __shfl_xor(rs, 1, 32);
      rs += __shfl_xor(rs, 2, 32);
      rs += __shfl_xor(rs, 4, 32);
      rs += __shfl_xor(rs, 8, 32);
      lrow[r] = lrow[r] * al + rs;
      mrow[r] = mnew;
      alpha[r] = al;
      pr0[r] = p0; pr1[r] = p1;
    }
#pragma unroll
    for (int t = 0; t < 4; ++t)
#pragma unroll
      for (int r = 0; r < 8; ++r) o[t][r] *= alpha[r];

    // P -> LDS (per-wave), re-read as A fragment
    unsigned short* P = Pw[wave];
#pragma unroll
    for (int r = 0; r < 8; ++r) {
      int m = half * 8 + r;
      P[m * 32 + ln16]      = f2bf(pr0[r]);
      P[m * 32 + 16 + ln16] = f2bf(pr1[r]);
    }
    asm volatile("s_wait_dscnt 0x0" ::: "memory");
    bf16x16 pa = load_fragA(&P[ln16 * 32]);
#pragma unroll
    for (int t = 0; t < 4; ++t) {
      bf16x16 bv = load_fragB(&Vtl[(t * 16 + ln16) * 32]);
      o[t] = wmma_bf16(pa, bv, o[t]);
    }
  }

  // Epilogue: divide by denom, fp32 out [B,S,H*Dh]
#pragma unroll
  for (int r = 0; r < 8; ++r) {
    int q = q0 + half * 8 + r;
    float inv = 1.0f / lrow[r];
    float* dst = out + ((size_t)(b * S_ + q)) * (H_ * DH_) + h * DH_;
#pragma unroll
    for (int t = 0; t < 4; ++t)
      dst[t * 16 + ln16] = o[t][r] * inv;
  }
}

extern "C" void kernel_launch(void* const* d_in, const int* in_sizes, int n_in,
                              void* d_out, int out_size, void* d_ws, size_t ws_size,
                              hipStream_t stream) {
  (void)in_sizes; (void)n_in; (void)out_size; (void)ws_size;
  const float* X    = (const float*)d_in[0];
  const float* W    = (const float*)d_in[1];
  const float* bias = (const float*)d_in[2];
  float* out = (float*)d_out;

  unsigned short* Qb  = (unsigned short*)d_ws;
  unsigned short* Kb  = Qb + QKV_ELEMS;
  unsigned short* Vt  = Kb + QKV_ELEMS;
  unsigned short* Xb  = Vt + QKV_ELEMS;
  unsigned short* Wt  = Xb + XB_ELEMS;

  prep_kernel<<<dim3(4096 + 3072), 256, 0, stream>>>(X, W, Xb, Wt);

  dim3 g1(N3_ / 128, (B_ * S_) / 128);     // 24 x 32
  qkv_gemm_kernel<<<g1, 256, 0, stream>>>(Xb, Wt, bias, Qb, Kb, Vt);

  dim3 g2(S_ / 64, H_, B_);                // 32 x 16 x 2
  flash_attn_kernel<<<g2, 128, 0, stream>>>(Qb, Kb, Vt, out);
}
